// BaselinePhasorBlock_37082747633877
// MI455X (gfx1250) — compile-verified, matrix operand order checked by reading, not run on hardware
//
#include <hip/hip_runtime.h>
#include <hip/hip_bf16.h>
#include <math.h>

// ---------------------------------------------------------------------------
// PhasorBlock on MI455X (gfx1250, wave32, WMMA).
// retrieved = tril(Phi_q @ Phi_k^T) @ V  (causal attention reformulation)
// All GEMMs: v_wmma_f32_16x16x32_bf16, 32x32 register-blocked tile per wave
// (4 wmma / k-step, 2x fragment reuse -> 16 FLOP/B of L2 traffic).
// ---------------------------------------------------------------------------

#define Dd 512
#define Kd 64
#define Bd 2
#define Ld 1024
#define Md (Bd * Ld)          // 2048 tokens
#define PHI_W (2 * Kd)        // 128
#define PI_F 3.14159265358979323846f

typedef __attribute__((ext_vector_type(16))) __bf16 v16bf;
typedef __attribute__((ext_vector_type(8)))  __bf16 v8bf;
typedef __attribute__((ext_vector_type(8)))  float  v8f;

union BF16x16 { v16bf v; v8bf h[2]; };

static __device__ __forceinline__ unsigned short f32_to_bf16(float f) {
  unsigned int u = __float_as_uint(f);
  u += 0x7FFFu + ((u >> 16) & 1u);   // round-to-nearest-even
  return (unsigned short)(u >> 16);
}

// ---------------------------------------------------------------------------
// Core: 32x32 f32 tile per wave via 4x v_wmma_f32_16x16x32_bf16 per k-step.
// A: row-major bf16 bits, lda elems. Bt[n][k] = B[k][n], ldb elems.
// ISA 16-bit A/B fragment layout: lanes 0-15 carry K {0..7,16..23},
// lanes 16-31 carry K {8..15,24..31} -> two b128 loads per fragment.
// acc index = mi*2 + ni (mi,ni in 0..1 -> 16x16 sub-tiles).
// ---------------------------------------------------------------------------
static __device__ __forceinline__ void wmma_tile32(
    const unsigned short* __restrict__ A, int lda,
    const unsigned short* __restrict__ Bt, int ldb,
    int kmax, int m0, int n0, int lane, v8f acc[4]) {
  const int half = (lane >> 4) & 1;
  const int idx  = lane & 15;
  const unsigned short* ap0 = A  + (size_t)(m0 + idx) * lda + half * 8;
  const unsigned short* ap1 = ap0 + (size_t)16 * lda;
  const unsigned short* bp0 = Bt + (size_t)(n0 + idx) * ldb + half * 8;
  const unsigned short* bp1 = bp0 + (size_t)16 * ldb;
  for (int k = 0; k < kmax; k += 32) {
    BF16x16 a0, a1, b0, b1;
    a0.h[0] = *(const v8bf*)(ap0 + k);
    a0.h[1] = *(const v8bf*)(ap0 + k + 16);
    a1.h[0] = *(const v8bf*)(ap1 + k);
    a1.h[1] = *(const v8bf*)(ap1 + k + 16);
    b0.h[0] = *(const v8bf*)(bp0 + k);
    b0.h[1] = *(const v8bf*)(bp0 + k + 16);
    b1.h[0] = *(const v8bf*)(bp1 + k);
    b1.h[1] = *(const v8bf*)(bp1 + k + 16);
    if (k + 32 < kmax) {  // global_prefetch_b8 one k-tile ahead
      __builtin_prefetch(ap0 + k + 32, 0, 3);
      __builtin_prefetch(ap1 + k + 32, 0, 3);
      __builtin_prefetch(bp0 + k + 32, 0, 3);
      __builtin_prefetch(bp1 + k + 32, 0, 3);
    }
    acc[0] = __builtin_amdgcn_wmma_f32_16x16x32_bf16(false, a0.v, false, b0.v, (short)0, acc[0], false, false);
    acc[1] = __builtin_amdgcn_wmma_f32_16x16x32_bf16(false, a0.v, false, b1.v, (short)0, acc[1], false, false);
    acc[2] = __builtin_amdgcn_wmma_f32_16x16x32_bf16(false, a1.v, false, b0.v, (short)0, acc[2], false, false);
    acc[3] = __builtin_amdgcn_wmma_f32_16x16x32_bf16(false, a1.v, false, b1.v, (short)0, acc[3], false, false);
  }
}

// C/D layout (per 16x16 sub-tile): VGPR r, lanes 0-15 -> M = r,
// lanes 16-31 -> M = 8 + r; N = lane & 15.
#define TILE_SETUP(Mtot, Ntot)                                  \
  const int lane = threadIdx.x & 31;                            \
  const int wave = threadIdx.x >> 5;                            \
  const int tile = blockIdx.x * 8 + wave;                       \
  const int tiles_n = (Ntot) / 32;                              \
  const int tm = tile / tiles_n;                                \
  const int tn = tile - tm * tiles_n;                           \
  if (tm >= (Mtot) / 32) return;                                \
  const int m0 = tm * 32, n0 = tn * 32;                         \
  const int half = (lane >> 4) & 1;                             \
  const int idx  = lane & 15;

// iterate the 4 accumulators: macro body sees (m, n, v)
#define EPILOGUE_FOREACH(acc, BODY)                             \
  _Pragma("unroll")                                             \
  for (int mi = 0; mi < 2; ++mi)                                \
    _Pragma("unroll")                                           \
    for (int ni = 0; ni < 2; ++ni)                              \
      _Pragma("unroll")                                         \
      for (int r = 0; r < 8; ++r) {                             \
        const int m = m0 + mi * 16 + half * 8 + r;              \
        const int n = n0 + ni * 16 + idx;                       \
        float v = acc[mi * 2 + ni][r];                          \
        BODY                                                    \
      }

// --------------------------- elementwise helpers ---------------------------

__global__ void cvt_bf16_kernel(const float* __restrict__ in,
                                unsigned short* __restrict__ out, int n) {
  int i = blockIdx.x * 256 + threadIdx.x;
  if (i < n) out[i] = f32_to_bf16(in[i]);
}

// W: rows x cols f32 (row-major) -> Wt: cols x rows bf16, Wt[n][k] = W[k][n]
__global__ void transpose_cvt_kernel(const float* __restrict__ W,
                                     unsigned short* __restrict__ Wt,
                                     int rows, int cols) {
  int i = blockIdx.x * 256 + threadIdx.x;
  if (i >= rows * cols) return;
  int k = i / cols, n = i - k * cols;
  Wt[(size_t)n * rows + k] = f32_to_bf16(W[i]);
}

// ------------------------------ GEMM kernels -------------------------------

// H = GELU(A @ W1 + b1), bf16 out
__global__ void gemm_gelu_kernel(const unsigned short* __restrict__ A,
                                 const unsigned short* __restrict__ Wt,
                                 const float* __restrict__ bias,
                                 unsigned short* __restrict__ H,
                                 int M, int N, int kdim) {
  TILE_SETUP(M, N)
  v8f acc[4] = {};
  wmma_tile32(A, kdim, Wt, kdim, kdim, m0, n0, lane, acc);
  EPILOGUE_FOREACH(acc, {
    v += bias[n];
    v = 0.5f * v * (1.0f + erff(v * 0.70710678118654752f));  // exact GELU
    H[(size_t)m * N + n] = f32_to_bf16(v);
  })
}

// ph = tanh(H @ W2 + b2) * pi; Phi[:, n] = cos(ph), Phi[:, 64+n] = sin(ph)
__global__ void gemm_phase_kernel(const unsigned short* __restrict__ H,
                                  const unsigned short* __restrict__ W2t,
                                  const float* __restrict__ b2,
                                  unsigned short* __restrict__ Phi,
                                  int M, int kdim) {
  TILE_SETUP(M, Kd)
  v8f acc[4] = {};
  wmma_tile32(H, kdim, W2t, kdim, kdim, m0, n0, lane, acc);
  EPILOGUE_FOREACH(acc, {
    float ph = tanhf(v + b2[n]) * PI_F;
    size_t row = (size_t)m * PHI_W;
    Phi[row + n]      = f32_to_bf16(cosf(ph));
    Phi[row + Kd + n] = f32_to_bf16(sinf(ph));
  })
}

// V = A @ Wv + bv, stored TRANSPOSED: Vt[d][token], ld = Md
__global__ void gemm_valueT_kernel(const unsigned short* __restrict__ A,
                                   const unsigned short* __restrict__ Wt,
                                   const float* __restrict__ bias,
                                   unsigned short* __restrict__ Vt,
                                   int M, int N, int kdim) {
  TILE_SETUP(M, N)
  v8f acc[4] = {};
  wmma_tile32(A, kdim, Wt, kdim, kdim, m0, n0, lane, acc);
  EPILOGUE_FOREACH(acc, {
    Vt[(size_t)n * Md + m] = f32_to_bf16(v + bias[n]);
  })
}

// S_b = tril(PhiQ_b @ PhiK_b^T), bf16, per batch (blockIdx.y)
__global__ void gemm_scores_kernel(const unsigned short* __restrict__ PhiQ,
                                   const unsigned short* __restrict__ PhiK,
                                   unsigned short* __restrict__ S) {
  const int bidx = blockIdx.y;
  TILE_SETUP(Ld, Ld)
  unsigned short* Sb = S + (size_t)bidx * Ld * Ld;
  if (n0 >= m0 + 32) {  // strictly above the diagonal: whole tile is zero
    v8f z[4] = {};
    EPILOGUE_FOREACH(z, {
      (void)v;
      Sb[(size_t)m * Ld + n] = 0;
    })
    return;
  }
  const unsigned short* Aq = PhiQ + (size_t)bidx * Ld * PHI_W;
  const unsigned short* Ak = PhiK + (size_t)bidx * Ld * PHI_W;
  v8f acc[4] = {};
  wmma_tile32(Aq, PHI_W, Ak, PHI_W, PHI_W, m0, n0, lane, acc);
  EPILOGUE_FOREACH(acc, {
    if (n > m) v = 0.0f;   // causal mask
    Sb[(size_t)m * Ld + n] = f32_to_bf16(v);
  })
}

// retrieved_b = S_b @ V_b  (K-loop truncated at t0+32 by causality), f32 out
__global__ void gemm_retr_kernel(const unsigned short* __restrict__ S,
                                 const unsigned short* __restrict__ Vt,
                                 float* __restrict__ retr) {
  const int bidx = blockIdx.y;
  TILE_SETUP(Ld, Dd)
  const unsigned short* Sb  = S  + (size_t)bidx * Ld * Ld;
  const unsigned short* Btv = Vt + (size_t)bidx * Ld;   // Vt[d][b*Ld + s], ld = Md
  int kmax = m0 + 32;                                   // S[t][s]=0 for s>t
  if (kmax > Ld) kmax = Ld;
  v8f acc[4] = {};
  wmma_tile32(Sb, Ld, Btv, Md, kmax, m0, n0, lane, acc);
  float* Rb = retr + (size_t)bidx * Ld * Dd;
  EPILOGUE_FOREACH(acc, {
    Rb[(size_t)m * Dd + n] = v;
  })
}

// out = x + Y_ln @ Wo + bo, f32 out
__global__ void gemm_out_kernel(const unsigned short* __restrict__ Y,
                                const unsigned short* __restrict__ Wot,
                                const float* __restrict__ bo,
                                const float* __restrict__ x,
                                float* __restrict__ out,
                                int M, int N, int kdim) {
  TILE_SETUP(M, N)
  v8f acc[4] = {};
  wmma_tile32(Y, kdim, Wot, kdim, kdim, m0, n0, lane, acc);
  EPILOGUE_FOREACH(acc, {
    size_t o = (size_t)m * N + n;
    out[o] = x[o] + v + bo[n];
  })
}

// ----------------------- LayerNorm(retrieved / norm) -----------------------
__global__ void ln_kernel(const float* __restrict__ retr,
                          const float* __restrict__ g,
                          const float* __restrict__ bb,
                          unsigned short* __restrict__ Y) {
  const int row = blockIdx.x;           // 0..Md-1
  const int t = row & (Ld - 1);         // position within batch
  const float scale = rsqrtf((float)(t + 1) * (float)Kd);
  __shared__ float s1[256], s2[256];
  const float* rp = retr + (size_t)row * Dd;
  const int tid = threadIdx.x;
  float v0 = rp[tid] * scale;
  float v1 = rp[tid + 256] * scale;
  s1[tid] = v0 + v1;
  s2[tid] = v0 * v0 + v1 * v1;
  __syncthreads();
  for (int off = 128; off > 0; off >>= 1) {
    if (tid < off) { s1[tid] += s1[tid + off]; s2[tid] += s2[tid + off]; }
    __syncthreads();
  }
  const float mu  = s1[0] * (1.0f / Dd);
  const float var = s2[0] * (1.0f / Dd) - mu * mu;
  const float inv = rsqrtf(var + 1e-5f);
  Y[(size_t)row * Dd + tid]       = f32_to_bf16((v0 - mu) * inv * g[tid] + bb[tid]);
  Y[(size_t)row * Dd + tid + 256] = f32_to_bf16((v1 - mu) * inv * g[tid + 256] + bb[tid + 256]);
}

// ------------------------------- launcher ----------------------------------

extern "C" void kernel_launch(void* const* d_in, const int* in_sizes, int n_in,
                              void* d_out, int out_size, void* d_ws, size_t ws_size,
                              hipStream_t stream) {
  const float* x    = (const float*)d_in[0];
  const float* Wk1  = (const float*)d_in[1];
  const float* bk1  = (const float*)d_in[2];
  const float* Wk2  = (const float*)d_in[3];
  const float* bk2  = (const float*)d_in[4];
  const float* Wq1  = (const float*)d_in[5];
  const float* bq1  = (const float*)d_in[6];
  const float* Wq2  = (const float*)d_in[7];
  const float* bq2  = (const float*)d_in[8];
  const float* Wv   = (const float*)d_in[9];
  const float* bv   = (const float*)d_in[10];
  const float* ln_g = (const float*)d_in[11];
  const float* ln_b = (const float*)d_in[12];
  const float* Wo   = (const float*)d_in[13];
  const float* bo   = (const float*)d_in[14];
  float* out = (float*)d_out;

  // workspace carve-up (bf16 buffers stored as ushort bit patterns)
  char* ws = (char*)d_ws;
  size_t off = 0;
  auto alloc = [&](size_t bytes) { char* p = ws + off; off += (bytes + 255) & ~(size_t)255; return p; };
  unsigned short* XB   = (unsigned short*)alloc((size_t)Md * Dd * 2);
  unsigned short* WK1T = (unsigned short*)alloc((size_t)Dd * Dd * 2);
  unsigned short* WQ1T = (unsigned short*)alloc((size_t)Dd * Dd * 2);
  unsigned short* WVT  = (unsigned short*)alloc((size_t)Dd * Dd * 2);
  unsigned short* WOT  = (unsigned short*)alloc((size_t)Dd * Dd * 2);
  unsigned short* WK2T = (unsigned short*)alloc((size_t)Kd * Dd * 2);
  unsigned short* WQ2T = (unsigned short*)alloc((size_t)Kd * Dd * 2);
  unsigned short* HK   = (unsigned short*)alloc((size_t)Md * Dd * 2);
  unsigned short* HQ   = (unsigned short*)alloc((size_t)Md * Dd * 2);
  unsigned short* PHIK = (unsigned short*)alloc((size_t)Md * PHI_W * 2);
  unsigned short* PHIQ = (unsigned short*)alloc((size_t)Md * PHI_W * 2);
  unsigned short* VT   = (unsigned short*)alloc((size_t)Dd * Md * 2);
  unsigned short* S    = (unsigned short*)alloc((size_t)Bd * Ld * Ld * 2);
  float*          RETR = (float*)alloc((size_t)Md * Dd * 4);
  unsigned short* YLN  = (unsigned short*)alloc((size_t)Md * Dd * 2);
  if (off > ws_size) return;  // workspace too small

  // 1) precision conversion / weight transposes
  cvt_bf16_kernel<<<(Md * Dd) / 256, 256, 0, stream>>>(x, XB, Md * Dd);
  transpose_cvt_kernel<<<(Dd * Dd) / 256, 256, 0, stream>>>(Wk1, WK1T, Dd, Dd);
  transpose_cvt_kernel<<<(Dd * Dd) / 256, 256, 0, stream>>>(Wq1, WQ1T, Dd, Dd);
  transpose_cvt_kernel<<<(Dd * Dd) / 256, 256, 0, stream>>>(Wv,  WVT,  Dd, Dd);
  transpose_cvt_kernel<<<(Dd * Dd) / 256, 256, 0, stream>>>(Wo,  WOT,  Dd, Dd);
  transpose_cvt_kernel<<<(Dd * Kd) / 256, 256, 0, stream>>>(Wk2, WK2T, Dd, Kd);
  transpose_cvt_kernel<<<(Dd * Kd) / 256, 256, 0, stream>>>(Wq2, WQ2T, Dd, Kd);

  // 2) encoders: hidden layers (GELU) then phase -> cos/sin features
  gemm_gelu_kernel<<<(Md / 32) * (Dd / 32) / 8, 256, 0, stream>>>(XB, WK1T, bk1, HK, Md, Dd, Dd);
  gemm_gelu_kernel<<<(Md / 32) * (Dd / 32) / 8, 256, 0, stream>>>(XB, WQ1T, bq1, HQ, Md, Dd, Dd);
  gemm_phase_kernel<<<(Md / 32) * (Kd / 32) / 8, 256, 0, stream>>>(HK, WK2T, bk2, PHIK, Md, Dd);
  gemm_phase_kernel<<<(Md / 32) * (Kd / 32) / 8, 256, 0, stream>>>(HQ, WQ2T, bq2, PHIQ, Md, Dd);

  // 3) values (written transposed for the S@V GEMM)
  gemm_valueT_kernel<<<(Md / 32) * (Dd / 32) / 8, 256, 0, stream>>>(XB, WVT, bv, VT, Md, Dd, Dd);

  // 4) causal attention: scores then retrieved
  dim3 gs((Ld / 32) * (Ld / 32) / 8, Bd);
  gemm_scores_kernel<<<gs, 256, 0, stream>>>(PHIQ, PHIK, S);
  dim3 gr((Ld / 32) * (Dd / 32) / 8, Bd);
  gemm_retr_kernel<<<gr, 256, 0, stream>>>(S, VT, RETR);

  // 5) scaled LayerNorm, then output projection + residual
  ln_kernel<<<Md, 256, 0, stream>>>(RETR, ln_g, ln_b, YLN);
  gemm_out_kernel<<<(Md / 32) * (Dd / 32) / 8, 256, 0, stream>>>(YLN, WOT, bo, x, out, Md, Dd, Dd);
  (void)in_sizes; (void)n_in; (void)out_size;
}
